// GraphTransformer_90237262889124
// MI455X (gfx1250) — matrix operand order repaired and checked by reading, not compiled, not run on hardware
//
#include <hip/hip_runtime.h>
#include <hip/hip_bf16.h>

// ---------------------------------------------------------------------------
// Informer-style encoder/decoder, one workgroup (8 wave32) per (b,n) sequence.
// All GEMMs via v_wmma_f32_16x16x32_f16, activations resident in LDS.
// ---------------------------------------------------------------------------

typedef __attribute__((ext_vector_type(16))) _Float16 v16h;
typedef __attribute__((ext_vector_type(8)))  _Float16 h8;
typedef __attribute__((ext_vector_type(8)))  float    v8f;

#define D_MODEL 64
#define L_ENC   96
#define L_DEC   144
#define OUT_LEN 48
#define N_NODES 325
#define BATCH   16

// ------------------------- shared memory layout (bytes) --------------------
#define SM_SMALL_OFF   0        // 512 f32: series[0..159], convw[160..351], mean[352], dmean[353]
#define SM_BIAS_OFF    2048     // 256 f32: biasA[0..63], biasB[64..127], gamma[128..191], beta[192..255]
#define SM_W_OFF       3072     // 64x64 f16  (current weight, stored [out][in])
#define SM_EMB_OFF     11264    // 96x64 f16  (embedding / residual)
#define SM_Q_OFF       23552    // 96x64 f16
#define SM_K_OFF       35840    // 96x64 f16  (stored [s][e] -> B operand for QK^T)
#define SM_VT_OFF      48128    // 64x96 f16  (V transposed -> B operand for P*V)
#define SM_S_OFF       60416    // 96x96 f16  (scores / FFN hidden)
#define SM_F_OFF       78848    // 96x64 f16
#define SM_G_OFF       91136    // 96x64 f16
#define SM_EO_OFF      103424   // 96x64 f16  (encoder output, persists)
#define SM_TOTAL       115712

struct Params {
  const float *x_enc, *x_dec, *enc_conv_w, *dec_conv_w;
  const float *eWq, *ebq, *eWk, *ebk, *eWv, *ebv, *eWo, *ebo;
  const float *ec1_w, *ec1_b, *ec2_w, *ec2_b;
  const float *en1_g, *en1_b, *en2_g, *en2_b;
  const float *dWq, *dbq, *dWk, *dbk, *dWv, *dbv, *dWo, *dbo;
  const float *dc1_w, *dc1_b, *dc2_w, *dc2_b;
  const float *dn2_g, *dn2_b, *dn3_g, *dn3_b;
  const float *proj_w, *proj_b;
  float *out;
};

// --------------------- WMMA tile loaders (CDNA5 §7.12.2) -------------------
// A tile 16x32 f16 from row-major [M][K] storage (ld in halfs).
__device__ __forceinline__ v16h load_a_tile(const _Float16* base, int ld, int lane) {
  int m    = lane & 15;
  int koff = (lane < 16) ? 0 : 8;
  const _Float16* p = base + m * ld + koff;
  h8 lo = *(const h8*)(p);        // K = koff .. koff+7
  h8 hi = *(const h8*)(p + 16);   // K = koff+16 .. koff+23
  v16h r;
#pragma unroll
  for (int i = 0; i < 8; ++i) { r[i] = lo[i]; r[i + 8] = hi[i]; }
  return r;
}

// B tile 32x16 f16 from TRANSPOSED storage Bt[N][K] (ld in halfs).
__device__ __forceinline__ v16h load_b_tile(const _Float16* base, int ld, int lane) {
  int nn = lane & 15;
  int k0 = (lane < 16) ? 0 : 8;
  const _Float16* p = base + nn * ld + k0;
  h8 lo = *(const h8*)(p);        // K = k0 .. k0+7
  h8 hi = *(const h8*)(p + 16);   // K = k0+16 .. k0+23
  v16h r;
#pragma unroll
  for (int i = 0; i < 8; ++i) { r[i] = lo[i]; r[i + 8] = hi[i]; }
  return r;
}

// ------------------------------- tiled GEMM --------------------------------
// C[M][Nt*16] (f16) = epilogue( scale * (A[M][K] @ Bt^T) + bias [+ res] )
// Bt stored [N][K].  transposedC writes C as [N][M] (ldc over M).
__device__ __forceinline__ void gemm_tiles(
    const _Float16* A, int lda, const _Float16* Bt, int ldb, int K,
    int Mtiles, int Ntiles, _Float16* C, int ldc,
    const float* bias, const _Float16* res, int ldres,
    float scale, bool relu, bool transposedC, int wv, int lane)
{
  int total = Mtiles * Ntiles;
  for (int t = wv; t < total; t += 8) {
    int mt = t / Ntiles, nt = t - mt * Ntiles;
    v8f acc = {};
    for (int kk = 0; kk < K; kk += 32) {
      v16h a = load_a_tile(A + (mt * 16) * lda + kk, lda, lane);
      v16h b = load_b_tile(Bt + (nt * 16) * ldb + kk, ldb, lane);
      acc = __builtin_amdgcn_wmma_f32_16x16x32_f16(false, a, false, b,
                                                   (short)0, acc, false, false);
    }
    int nn = lane & 15;
    int mh = (lane >> 4) << 3;                 // 0 or 8
    float bv = bias ? bias[nt * 16 + nn] : 0.f;
    if (!transposedC) {
#pragma unroll
      for (int r = 0; r < 8; ++r) {
        float v = acc[r] * scale + bv;
        if (res)  v += (float)res[(mt * 16 + mh + r) * ldres + nt * 16 + nn];
        if (relu) v = v > 0.f ? v : 0.f;
        C[(mt * 16 + mh + r) * ldc + nt * 16 + nn] = (_Float16)v;
      }
    } else {
      h8 pk;
#pragma unroll
      for (int r = 0; r < 8; ++r) pk[r] = (_Float16)(acc[r] * scale + bv);
      *(h8*)(C + (nt * 16 + nn) * ldc + mt * 16 + mh) = pk;  // C[n][m]
    }
  }
}

// Load a 64x64 weight slice, transposed to [out][in] f16 in LDS.
// W is global [64][cols]; takes columns [ocol0, ocol0+64).
__device__ __forceinline__ void load_w_t(const float* W, int cols, int ocol0,
                                         _Float16* wh, int tid) {
  for (int idx = tid; idx < 64 * 64; idx += 256) {
    int i = idx >> 6, o = idx & 63;
    wh[o * 64 + i] = (_Float16)W[i * cols + ocol0 + o];
  }
}

__device__ __forceinline__ void layer_norm(const _Float16* src, _Float16* dst,
                                           int M, const float* g, const float* b, int tid) {
  for (int t = tid; t < M; t += 256) {
    const _Float16* r = src + t * 64;
    float s = 0.f;
    for (int c = 0; c < 64; ++c) s += (float)r[c];
    float mu = s * (1.f / 64.f), s2 = 0.f;
    for (int c = 0; c < 64; ++c) { float v = (float)r[c] - mu; s2 += v * v; }
    float inv = rsqrtf(s2 * (1.f / 64.f) + 1e-5f);
    _Float16* w = dst + t * 64;
    for (int c = 0; c < 64; ++c) {
      float v = ((float)r[c] - mu) * inv;
      w[c] = (_Float16)(v * g[c] + b[c]);
    }
  }
}

__device__ __forceinline__ void softmax_rows(_Float16* S, int M, int ld, int tid) {
  for (int t = tid; t < M; t += 256) {
    _Float16* r = S + t * ld;
    float mx = -1e30f;
    for (int c = 0; c < 96; ++c) mx = fmaxf(mx, (float)r[c]);
    float sum = 0.f;
    for (int c = 0; c < 96; ++c) sum += __expf((float)r[c] - mx);
    float inv = 1.f / sum;
    for (int c = 0; c < 96; ++c) r[c] = (_Float16)(__expf((float)r[c] - mx) * inv);
  }
}

// ---------------------------------------------------------------------------
__global__ void __launch_bounds__(256, 1)
informer_seq_kernel(Params p) {
  extern __shared__ char smem[];
  float*    sm_small = (float*)(smem + SM_SMALL_OFF);
  float*    sm_bias  = (float*)(smem + SM_BIAS_OFF);
  _Float16* sm_w   = (_Float16*)(smem + SM_W_OFF);
  _Float16* sm_emb = (_Float16*)(smem + SM_EMB_OFF);
  _Float16* sm_q   = (_Float16*)(smem + SM_Q_OFF);
  _Float16* sm_k   = (_Float16*)(smem + SM_K_OFF);
  _Float16* sm_vT  = (_Float16*)(smem + SM_VT_OFF);
  _Float16* sm_s   = (_Float16*)(smem + SM_S_OFF);
  _Float16* sm_f   = (_Float16*)(smem + SM_F_OFF);
  _Float16* sm_g   = (_Float16*)(smem + SM_G_OFF);
  _Float16* sm_eo  = (_Float16*)(smem + SM_EO_OFF);

  const int tid  = threadIdx.x;
  const int wv   = tid >> 5;
  const int lane = tid & 31;
  const int b = blockIdx.x / N_NODES;
  const int n = blockIdx.x - b * N_NODES;

  // ============================ ENCODER =====================================
  // series + conv weights
  for (int t = tid; t < L_ENC; t += 256)
    sm_small[t] = p.x_enc[(b * L_ENC + t) * N_NODES + n];
  for (int i = tid; i < 192; i += 256) sm_small[160 + i] = p.enc_conv_w[i];
  __syncthreads();
  if (tid == 0) {
    float s = 0.f;
    for (int t = 0; t < L_ENC; ++t) s += sm_small[t];
    sm_small[352] = s * (1.f / L_ENC);
  }
  __syncthreads();
  const float menc = sm_small[352];

  // circular conv (k=3, Cin=1) -> emb f16 [96][64]
  for (int idx = tid; idx < L_ENC * 64; idx += 256) {
    int t = idx >> 6, c = idx & 63;
    const float* w = sm_small + 160 + c * 3;
    float xm1 = sm_small[(t + L_ENC - 1) % L_ENC] - menc;
    float x0  = sm_small[t] - menc;
    float xp1 = sm_small[(t + 1) % L_ENC] - menc;
    sm_emb[idx] = (_Float16)(w[0] * xm1 + w[1] * x0 + w[2] * xp1);
  }
  __syncthreads();

  // Q = emb @ eWq + bq
  load_w_t(p.eWq, 64, 0, sm_w, tid);
  if (tid < 64) sm_bias[tid] = p.ebq[tid];
  __syncthreads();
  gemm_tiles(sm_emb, 64, sm_w, 64, 64, 6, 4, sm_q, 64, sm_bias, nullptr, 0, 1.f, false, false, wv, lane);
  __syncthreads();
  // K
  load_w_t(p.eWk, 64, 0, sm_w, tid);
  if (tid < 64) sm_bias[tid] = p.ebk[tid];
  __syncthreads();
  gemm_tiles(sm_emb, 64, sm_w, 64, 64, 6, 4, sm_k, 64, sm_bias, nullptr, 0, 1.f, false, false, wv, lane);
  __syncthreads();
  // V (stored transposed [64][96])
  load_w_t(p.eWv, 64, 0, sm_w, tid);
  if (tid < 64) sm_bias[tid] = p.ebv[tid];
  __syncthreads();
  gemm_tiles(sm_emb, 64, sm_w, 64, 64, 6, 4, sm_vT, 96, sm_bias, nullptr, 0, 1.f, false, true, wv, lane);
  __syncthreads();

  // scores = (Q K^T)/8 ; softmax ; O = P V
  gemm_tiles(sm_q, 64, sm_k, 64, 64, 6, 6, sm_s, 96, nullptr, nullptr, 0, 0.125f, false, false, wv, lane);
  __syncthreads();
  softmax_rows(sm_s, L_ENC, 96, tid);
  __syncthreads();
  gemm_tiles(sm_s, 96, sm_vT, 96, 96, 6, 4, sm_f, 64, nullptr, nullptr, 0, 1.f, false, false, wv, lane);
  __syncthreads();

  // x = LN(emb + O@Wo + bo)
  load_w_t(p.eWo, 64, 0, sm_w, tid);
  if (tid < 64) sm_bias[tid] = p.ebo[tid];
  __syncthreads();
  gemm_tiles(sm_f, 64, sm_w, 64, 64, 6, 4, sm_g, 64, sm_bias, sm_emb, 64, 1.f, false, false, wv, lane);
  __syncthreads();
  if (tid < 64) { sm_bias[128 + tid] = p.en1_g[tid]; sm_bias[192 + tid] = p.en1_b[tid]; }
  __syncthreads();
  layer_norm(sm_g, sm_g, L_ENC, sm_bias + 128, sm_bias + 192, tid);
  __syncthreads();

  // FFN (d_ff == d): enc_out = LN(x + relu(x@W1+b1)@W2+b2)
  load_w_t(p.ec1_w, 64, 0, sm_w, tid);
  if (tid < 64) sm_bias[tid] = p.ec1_b[tid];
  __syncthreads();
  gemm_tiles(sm_g, 64, sm_w, 64, 64, 6, 4, sm_s, 64, sm_bias, nullptr, 0, 1.f, true, false, wv, lane);
  __syncthreads();
  load_w_t(p.ec2_w, 64, 0, sm_w, tid);
  if (tid < 64) sm_bias[tid] = p.ec2_b[tid];
  __syncthreads();
  gemm_tiles(sm_s, 64, sm_w, 64, 64, 6, 4, sm_f, 64, sm_bias, sm_g, 64, 1.f, false, false, wv, lane);
  __syncthreads();
  if (tid < 64) { sm_bias[128 + tid] = p.en2_g[tid]; sm_bias[192 + tid] = p.en2_b[tid]; }
  __syncthreads();
  layer_norm(sm_f, sm_eo, L_ENC, sm_bias + 128, sm_bias + 192, tid);
  __syncthreads();

  // ============================ DECODER =====================================
  // Only the last OUT_LEN rows are ever emitted, and all post-conv decoder ops
  // are row-local -> compute only rows [96,144) (local M = 48, 3 row tiles).
  for (int t = tid; t < L_DEC; t += 256)
    sm_small[t] = p.x_dec[(b * L_DEC + t) * N_NODES + n];
  for (int i = tid; i < 192; i += 256) sm_small[160 + i] = p.dec_conv_w[i];
  __syncthreads();
  if (tid == 0) {
    float s = 0.f;
    for (int t = 0; t < OUT_LEN; ++t) s += sm_small[t];
    sm_small[353] = s * (1.f / OUT_LEN);
  }
  __syncthreads();
  {
    float dmean = sm_small[353];
    for (int t = tid; t < OUT_LEN; t += 256) sm_small[t] -= dmean;
  }
  __syncthreads();
  for (int idx = tid; idx < OUT_LEN * 64; idx += 256) {
    int tr = 96 + (idx >> 6), c = idx & 63;
    const float* w = sm_small + 160 + c * 3;
    float xm1 = sm_small[tr - 1];
    float x0  = sm_small[tr];
    float xp1 = sm_small[(tr + 1) % L_DEC];
    sm_emb[idx] = (_Float16)(w[0] * xm1 + w[1] * x0 + w[2] * xp1);
  }
  __syncthreads();

  // Q from dec_emb (48 rows); K,V from enc_out (96 rows)
  load_w_t(p.dWq, 64, 0, sm_w, tid);
  if (tid < 64) sm_bias[tid] = p.dbq[tid];
  __syncthreads();
  gemm_tiles(sm_emb, 64, sm_w, 64, 64, 3, 4, sm_q, 64, sm_bias, nullptr, 0, 1.f, false, false, wv, lane);
  __syncthreads();
  load_w_t(p.dWk, 64, 0, sm_w, tid);
  if (tid < 64) sm_bias[tid] = p.dbk[tid];
  __syncthreads();
  gemm_tiles(sm_eo, 64, sm_w, 64, 64, 6, 4, sm_k, 64, sm_bias, nullptr, 0, 1.f, false, false, wv, lane);
  __syncthreads();
  load_w_t(p.dWv, 64, 0, sm_w, tid);
  if (tid < 64) sm_bias[tid] = p.dbv[tid];
  __syncthreads();
  gemm_tiles(sm_eo, 64, sm_w, 64, 64, 6, 4, sm_vT, 96, sm_bias, nullptr, 0, 1.f, false, true, wv, lane);
  __syncthreads();

  gemm_tiles(sm_q, 64, sm_k, 64, 64, 3, 6, sm_s, 96, nullptr, nullptr, 0, 0.125f, false, false, wv, lane);
  __syncthreads();
  softmax_rows(sm_s, OUT_LEN, 96, tid);
  __syncthreads();
  gemm_tiles(sm_s, 96, sm_vT, 96, 96, 3, 4, sm_f, 64, nullptr, nullptr, 0, 1.f, false, false, wv, lane);
  __syncthreads();

  load_w_t(p.dWo, 64, 0, sm_w, tid);
  if (tid < 64) sm_bias[tid] = p.dbo[tid];
  __syncthreads();
  gemm_tiles(sm_f, 64, sm_w, 64, 64, 3, 4, sm_g, 64, sm_bias, sm_emb, 64, 1.f, false, false, wv, lane);
  __syncthreads();
  if (tid < 64) { sm_bias[128 + tid] = p.dn2_g[tid]; sm_bias[192 + tid] = p.dn2_b[tid]; }
  __syncthreads();
  layer_norm(sm_g, sm_g, OUT_LEN, sm_bias + 128, sm_bias + 192, tid);
  __syncthreads();

  // FFN d_ff = 256, chunked over 4 groups of 64; output tiles live in regs.
  if (tid < 64) sm_bias[64 + tid] = p.dc2_b[tid];
  v8f accA = {}, accB = {};
  const int t0 = wv, t1 = wv + 8;          // 12 output tiles (3x4)
  for (int g = 0; g < 4; ++g) {
    load_w_t(p.dc1_w, 256, g * 64, sm_w, tid);
    if (tid < 64) sm_bias[tid] = p.dc1_b[g * 64 + tid];
    __syncthreads();
    gemm_tiles(sm_g, 64, sm_w, 64, 64, 3, 4, sm_s, 64, sm_bias, nullptr, 0, 1.f, true, false, wv, lane);
    __syncthreads();
    load_w_t(p.dc2_w + g * 64 * 64, 64, 0, sm_w, tid);
    __syncthreads();
    {
      int mt = t0 >> 2, nt = t0 & 3;
      for (int kk = 0; kk < 64; kk += 32) {
        v16h a = load_a_tile(sm_s + mt * 16 * 64 + kk, 64, lane);
        v16h bb = load_b_tile(sm_w + nt * 16 * 64 + kk, 64, lane);
        accA = __builtin_amdgcn_wmma_f32_16x16x32_f16(false, a, false, bb, (short)0, accA, false, false);
      }
    }
    if (t1 < 12) {
      int mt = t1 >> 2, nt = t1 & 3;
      for (int kk = 0; kk < 64; kk += 32) {
        v16h a = load_a_tile(sm_s + mt * 16 * 64 + kk, 64, lane);
        v16h bb = load_b_tile(sm_w + nt * 16 * 64 + kk, 64, lane);
        accB = __builtin_amdgcn_wmma_f32_16x16x32_f16(false, a, false, bb, (short)0, accB, false, false);
      }
    }
    __syncthreads();
  }
  {
    int nn = lane & 15, mh = (lane >> 4) << 3;
    {
      int mt = t0 >> 2, nt = t0 & 3;
#pragma unroll
      for (int r = 0; r < 8; ++r) {
        int row = mt * 16 + mh + r, col = nt * 16 + nn;
        float v = accA[r] + sm_bias[64 + col] + (float)sm_g[row * 64 + col];
        sm_f[row * 64 + col] = (_Float16)v;
      }
    }
    if (t1 < 12) {
      int mt = t1 >> 2, nt = t1 & 3;
#pragma unroll
      for (int r = 0; r < 8; ++r) {
        int row = mt * 16 + mh + r, col = nt * 16 + nn;
        float v = accB[r] + sm_bias[64 + col] + (float)sm_g[row * 64 + col];
        sm_f[row * 64 + col] = (_Float16)v;
      }
    }
  }
  __syncthreads();
  if (tid < 64) { sm_bias[128 + tid] = p.dn3_g[tid]; sm_bias[192 + tid] = p.dn3_b[tid]; }
  __syncthreads();
  layer_norm(sm_f, sm_g, OUT_LEN, sm_bias + 128, sm_bias + 192, tid);
  __syncthreads();

  // final projection (64 -> 1) + enc mean, write (B, 48, 1, N)
  const float pb = p.proj_b[0];
  for (int t = tid; t < OUT_LEN; t += 256) {
    float s = pb;
    for (int c = 0; c < 64; ++c) s += (float)sm_g[t * 64 + c] * p.proj_w[c];
    p.out[(b * OUT_LEN + t) * N_NODES + n] = s + menc;
  }
}

// ---------------------------------------------------------------------------
extern "C" void kernel_launch(void* const* d_in, const int* in_sizes, int n_in,
                              void* d_out, int out_size, void* d_ws, size_t ws_size,
                              hipStream_t stream) {
  (void)in_sizes; (void)n_in; (void)d_ws; (void)ws_size; (void)out_size;
  Params p;
  p.x_enc      = (const float*)d_in[0];
  p.x_dec      = (const float*)d_in[1];
  // d_in[2] = adj_matrix (unused by reference), d_in[3] = out_len (fixed 48)
  p.enc_conv_w = (const float*)d_in[4];
  p.dec_conv_w = (const float*)d_in[5];
  p.eWq  = (const float*)d_in[6];   p.ebq = (const float*)d_in[7];
  p.eWk  = (const float*)d_in[8];   p.ebk = (const float*)d_in[9];
  p.eWv  = (const float*)d_in[10];  p.ebv = (const float*)d_in[11];
  p.eWo  = (const float*)d_in[12];  p.ebo = (const float*)d_in[13];
  p.ec1_w = (const float*)d_in[14]; p.ec1_b = (const float*)d_in[15];
  p.ec2_w = (const float*)d_in[16]; p.ec2_b = (const float*)d_in[17];
  p.en1_g = (const float*)d_in[18]; p.en1_b = (const float*)d_in[19];
  p.en2_g = (const float*)d_in[20]; p.en2_b = (const float*)d_in[21];
  p.dWq  = (const float*)d_in[22];  p.dbq = (const float*)d_in[23];
  p.dWk  = (const float*)d_in[24];  p.dbk = (const float*)d_in[25];
  p.dWv  = (const float*)d_in[26];  p.dbv = (const float*)d_in[27];
  p.dWo  = (const float*)d_in[28];  p.dbo = (const float*)d_in[29];
  p.dc1_w = (const float*)d_in[30]; p.dc1_b = (const float*)d_in[31];
  p.dc2_w = (const float*)d_in[32]; p.dc2_b = (const float*)d_in[33];
  p.dn2_g = (const float*)d_in[34]; p.dn2_b = (const float*)d_in[35];
  p.dn3_g = (const float*)d_in[36]; p.dn3_b = (const float*)d_in[37];
  p.proj_w = (const float*)d_in[38]; p.proj_b = (const float*)d_in[39];
  p.out = (float*)d_out;

  dim3 grid(BATCH * N_NODES);   // 5200 sequences, one workgroup each
  dim3 block(256);              // 8 wave32
  hipLaunchKernelGGL(informer_seq_kernel, grid, block, SM_TOTAL, stream, p);
}
